// SRN_model_36773509988382
// MI455X (gfx1250) — compile-verified, matrix operand order checked by reading, not run on hardware
//
#include <hip/hip_runtime.h>
#include <hip/hip_bf16.h>
#include <math.h>

typedef __attribute__((ext_vector_type(16))) __bf16 v16bf;
typedef __attribute__((ext_vector_type(8)))  float  v8f;

#define B_   64
#define S_   2048
#define NI_  256
#define NH_  512
#define NO_  256
#define KTOT_ 768            // NH + NI : fused [h | x] @ [U ; W]
#define NWG_  8              // recurrence workgroups (co-resident, 64 cols each)

// ---- workspace layout (bytes) ----
constexpr size_t kWS_UW     = 0;                            // [512][768] bf16 = 768 KB
constexpr size_t kWS_VC     = (size_t)512 * 768 * 2;        // [256][512] bf16 = 256 KB
constexpr size_t kWS_H0     = kWS_VC + (size_t)256 * 512 * 2;   // h ping (64 KB)
constexpr size_t kWS_H1     = kWS_H0 + (size_t)64 * 512 * 2;    // h pong (64 KB)
constexpr size_t kWS_CTR    = kWS_H1 + (size_t)64 * 512 * 2;    // barrier ctr (64 B)
constexpr size_t kWS_XT     = kWS_CTR + 64;                 // x bf16 (S,B,NI) = 64 MB
constexpr size_t kXT_BYTES  = (size_t)S_ * B_ * NI_ * 2;
constexpr size_t kWS_HSEQ   = kWS_XT + kXT_BYTES;           // hseq bf16 (S+1,B,NH) = 128 MB
constexpr size_t kHSEQ_BYTES = (size_t)(S_ + 1) * B_ * NH_ * 2;

__device__ __forceinline__ v16bf cvt16(float4 f0, float4 f1, float4 f2, float4 f3) {
    v16bf a;
    a[0]  = (__bf16)f0.x; a[1]  = (__bf16)f0.y; a[2]  = (__bf16)f0.z; a[3]  = (__bf16)f0.w;
    a[4]  = (__bf16)f1.x; a[5]  = (__bf16)f1.y; a[6]  = (__bf16)f1.z; a[7]  = (__bf16)f1.w;
    a[8]  = (__bf16)f2.x; a[9]  = (__bf16)f2.y; a[10] = (__bf16)f2.z; a[11] = (__bf16)f2.w;
    a[12] = (__bf16)f3.x; a[13] = (__bf16)f3.y; a[14] = (__bf16)f3.z; a[15] = (__bf16)f3.w;
    return a;
}

// ------------------------------------------------------------------
// Kernel 1: pack weights to bf16 (column-major along K), init h0, zero barrier
// ------------------------------------------------------------------
__global__ void srn_prep(const float* __restrict__ H0, const float* __restrict__ W,
                         const float* __restrict__ U,  const float* __restrict__ V,
                         char* __restrict__ ws, __bf16* __restrict__ h0dst) {
    __bf16* UW = (__bf16*)(ws + kWS_UW);
    __bf16* Vc = (__bf16*)(ws + kWS_VC);
    const int nUW = 512 * KTOT_;
    const int nVc = NO_ * NH_;
    const int nH0 = B_ * NH_;
    int idx = blockIdx.x * blockDim.x + threadIdx.x;
    if (idx < nUW) {
        int c = idx / KTOT_, k = idx % KTOT_;
        float v = (k < NH_) ? U[(size_t)k * NH_ + c] : W[(size_t)(k - NH_) * NH_ + c];
        UW[idx] = (__bf16)v;
    } else if (idx < nUW + nVc) {
        int j = idx - nUW;
        int c = j / NH_, k = j % NH_;
        Vc[j] = (__bf16)V[(size_t)k * NO_ + c];
    } else if (idx < nUW + nVc + nH0) {
        int j = idx - nUW - nVc;
        h0dst[j] = (__bf16)tanhf(H0[j % NH_]);
    } else if (idx == nUW + nVc + nH0) {
        *(unsigned int*)(ws + kWS_CTR) = 0u;
    }
}

// ------------------------------------------------------------------
// Kernel 1b (optional): x (B,S,NI) fp32 -> xT (S,B,NI) bf16.
// Each step's x tile becomes one contiguous 32 KB block on the serial path.
// ------------------------------------------------------------------
__global__ void srn_xpose(const float* __restrict__ x, char* __restrict__ ws) {
    __bf16* xT = (__bf16*)(ws + kWS_XT);
    size_t idx = (size_t)blockIdx.x * blockDim.x + threadIdx.x;
    const size_t total = (size_t)S_ * B_ * NI_;
    if (idx >= total) return;
    int k = (int)(idx % NI_);
    size_t r = idx / NI_;
    int b = (int)(r % B_);
    int t = (int)(r / B_);
    xT[idx] = (__bf16)x[((size_t)b * S_ + t) * NI_ + k];
}

// ------------------------------------------------------------------
// Kernel 2: sequential recurrence, weight-stationary in LDS.
//   MODE 0: x fp32 cvt on the fly, h ping-pong scratch
//   MODE 1: x bf16 (xT),            h ping-pong scratch
//   MODE 2: x bf16 (xT),            h written into full hseq (consumed by srn_out)
// ------------------------------------------------------------------
template <int MODE>
__global__ void __launch_bounds__(256, 1)
srn_recur(const float* __restrict__ x, const __bf16* __restrict__ xT,
          __bf16* __restrict__ hseq, const float* __restrict__ hid_bias,
          char* __restrict__ ws, float* __restrict__ hidden_out) {
    extern __shared__ char smem[];
    __bf16* ldsB = (__bf16*)smem;                 // [64 cols][KTOT_] bf16 = 96 KB

    const int wg   = blockIdx.x;
    const int tid  = threadIdx.x;
    const int lane = tid & 31;
    const int wave = tid >> 5;
    const int mt      = wave >> 1;                // 4 M-tiles of 16 rows
    const int ntBase  = (wave & 1) * 2;           // 2 N-tiles per wave
    const int rsel    = lane >> 4;
    const int lcol    = lane & 15;

    const __bf16* UW = (const __bf16*)(ws + kWS_UW);
    __bf16* hA = (__bf16*)(ws + kWS_H0);
    __bf16* hB = (__bf16*)(ws + kWS_H1);
    unsigned int* ctr = (unsigned int*)(ws + kWS_CTR);

    // stage this WG's [U;W] column slice into LDS (96 KB)
    {
        const float4* src = (const float4*)(UW + (size_t)wg * 64 * KTOT_);
        float4* dst = (float4*)ldsB;
        const int nvec = 64 * KTOT_ * 2 / 16;     // 6144
        for (int i = tid; i < nvec; i += 256) dst[i] = src[i];
    }
    __syncthreads();

    const int gc0 = wg * 64 + (ntBase + 0) * 16 + lcol;
    const int gc1 = gc0 + 16;
    const float bias0 = hid_bias[gc0];
    const float bias1 = hid_bias[gc1];
    const int arow = mt * 16 + lcol;

    for (int t = 0; t < S_; ++t) {
        const __bf16* hcur;
        __bf16* hnext;
        if (MODE == 2) {
            hcur  = hseq + (size_t)t * (B_ * NH_);
            hnext = hseq + (size_t)(t + 1) * (B_ * NH_);
        } else {
            hcur  = (t & 1) ? hB : hA;
            hnext = (t & 1) ? hA : hB;
        }

        // prefetch next step's x tile (hidden behind WMMAs + barrier)
        if (t + 1 < S_) {
            if (MODE >= 1) {
                const __bf16* p = xT + ((size_t)(t + 1) * B_ + arow) * NI_;
                #pragma unroll
                for (int ki = 0; ki < NI_ / 32; ++ki)
                    __builtin_prefetch(p + ki * 32 + rsel * 16, 0, 1);
            } else {
                const float* p = x + ((size_t)arow * S_ + (t + 1)) * NI_;
                #pragma unroll
                for (int ki = 0; ki < NI_ / 32; ++ki)
                    __builtin_prefetch(p + ki * 32 + rsel * 16, 0, 1);
            }
        }

        v8f acc0 = {}; v8f acc1 = {};

        // K = 0..511 : h_{t-1} @ U   (A already bf16 in L2-resident scratch)
        #pragma unroll 4
        for (int ki = 0; ki < NH_ / 32; ++ki) {
            const int koff = ki * 32 + rsel * 16;
            v16bf a = *(const v16bf*)(hcur + arow * NH_ + koff);
            v16bf b0 = *(const v16bf*)(ldsB + ((ntBase + 0) * 16 + lcol) * KTOT_ + koff);
            v16bf b1 = *(const v16bf*)(ldsB + ((ntBase + 1) * 16 + lcol) * KTOT_ + koff);
            acc0 = __builtin_amdgcn_wmma_f32_16x16x32_bf16(false, a, false, b0, (short)0, acc0, false, false);
            acc1 = __builtin_amdgcn_wmma_f32_16x16x32_bf16(false, a, false, b1, (short)0, acc1, false, false);
        }
        // K = 512..767 : x_t @ W
        #pragma unroll 4
        for (int ki = NH_ / 32; ki < KTOT_ / 32; ++ki) {
            const int koff = ki * 32 + rsel * 16;
            v16bf a;
            if (MODE >= 1) {
                a = *(const v16bf*)(xT + ((size_t)t * B_ + arow) * NI_ + (koff - NH_));
            } else {
                const float4* xp = (const float4*)(x + ((size_t)arow * S_ + t) * NI_ + (koff - NH_));
                a = cvt16(xp[0], xp[1], xp[2], xp[3]);
            }
            v16bf b0 = *(const v16bf*)(ldsB + ((ntBase + 0) * 16 + lcol) * KTOT_ + koff);
            v16bf b1 = *(const v16bf*)(ldsB + ((ntBase + 1) * 16 + lcol) * KTOT_ + koff);
            acc0 = __builtin_amdgcn_wmma_f32_16x16x32_bf16(false, a, false, b0, (short)0, acc0, false, false);
            acc1 = __builtin_amdgcn_wmma_f32_16x16x32_bf16(false, a, false, b1, (short)0, acc1, false, false);
        }

        // epilogue: bias + tanh, write fp32 hidden_seq and bf16 h_next
        const int row0 = mt * 16 + 8 * rsel;
        #pragma unroll
        for (int r = 0; r < 8; ++r) {
            const int grow = row0 + r;                          // batch index
            float v0 = tanhf(acc0[r] + bias0);
            float v1 = tanhf(acc1[r] + bias1);
            size_t obase = ((size_t)grow * S_ + t) * NH_;
            hidden_out[obase + gc0] = v0;
            hidden_out[obase + gc1] = v1;
            hnext[grow * NH_ + gc0] = (__bf16)v0;
            hnext[grow * NH_ + gc1] = (__bf16)v1;
        }

        // ---- device-scope step barrier ----
        __threadfence();
        __syncthreads();
        if (tid == 0) {
            __hip_atomic_fetch_add(ctr, 1u, __ATOMIC_ACQ_REL, __HIP_MEMORY_SCOPE_AGENT);
            const unsigned int target = (unsigned int)(t + 1) * NWG_;
            while (__hip_atomic_load(ctr, __ATOMIC_ACQUIRE, __HIP_MEMORY_SCOPE_AGENT) < target) {
                __builtin_amdgcn_s_sleep(1);
            }
        }
        __syncthreads();
    }
}

// ------------------------------------------------------------------
// Kernel 3: output GEMM  out = hidden_seq @ V + out_bias
//   BF16A=1: A read directly from bf16 hseq (t-major, coalesced, no cvt)
//   BF16A=0: A read from fp32 hidden in d_out, cvt on the fly
//   grid (2048, 4), block 256 = 8 waves, 64x64 C block, V staged in 64 KB LDS
// ------------------------------------------------------------------
template <bool BF16A>
__global__ void __launch_bounds__(256, 2)
srn_out(const float* __restrict__ hidden, const float* __restrict__ out_bias,
        const char* __restrict__ ws, float* __restrict__ out) {
    extern __shared__ char smem[];
    __bf16* ldsV = (__bf16*)smem;                 // [64 cols][NH_] bf16 = 64 KB
    const __bf16* Vc = (const __bf16*)(ws + kWS_VC);

    const int rb = blockIdx.x;                    // BF16A: t index; else 64-row block
    const int cb = blockIdx.y;                    // 64-col block of NO
    const int tid = threadIdx.x;
    const int lane = tid & 31, wave = tid >> 5;
    const int mt = wave >> 1, ntBase = (wave & 1) * 2;
    const int rsel = lane >> 4, lcol = lane & 15;

    {
        const float4* src = (const float4*)(Vc + (size_t)cb * 64 * NH_);
        float4* dst = (float4*)ldsV;
        const int nvec = 64 * NH_ * 2 / 16;       // 4096
        for (int i = tid; i < nvec; i += 256) dst[i] = src[i];
    }
    __syncthreads();

    const int arow = mt * 16 + lcol;
    const __bf16* hs = (const __bf16*)(ws + kWS_HSEQ);

    v8f acc0 = {}; v8f acc1 = {};
    #pragma unroll 4
    for (int ki = 0; ki < NH_ / 32; ++ki) {
        const int koff = ki * 32 + rsel * 16;
        v16bf a;
        if (BF16A) {
            // hseq slot (t+1) holds h_t ; rows are batch indices
            a = *(const v16bf*)(hs + ((size_t)(rb + 1) * B_ + arow) * NH_ + koff);
        } else {
            const float4* hp = (const float4*)(hidden + ((size_t)rb * 64 + arow) * NH_ + koff);
            a = cvt16(hp[0], hp[1], hp[2], hp[3]);
        }
        v16bf b0 = *(const v16bf*)(ldsV + ((ntBase + 0) * 16 + lcol) * NH_ + koff);
        v16bf b1 = *(const v16bf*)(ldsV + ((ntBase + 1) * 16 + lcol) * NH_ + koff);
        acc0 = __builtin_amdgcn_wmma_f32_16x16x32_bf16(false, a, false, b0, (short)0, acc0, false, false);
        acc1 = __builtin_amdgcn_wmma_f32_16x16x32_bf16(false, a, false, b1, (short)0, acc1, false, false);
    }

    const int gc0 = cb * 64 + (ntBase + 0) * 16 + lcol;
    const int gc1 = gc0 + 16;
    const float bias0 = out_bias[gc0];
    const float bias1 = out_bias[gc1];
    const int row0 = mt * 16 + 8 * rsel;
    #pragma unroll
    for (int r = 0; r < 8; ++r) {
        size_t obase;
        if (BF16A) {
            // row = batch b, column block fixed t = rb : out[(b*S + t)*NO + c]
            obase = ((size_t)(row0 + r) * S_ + rb) * NO_;
        } else {
            obase = ((size_t)rb * 64 + row0 + r) * NO_;
        }
        out[obase + gc0] = acc0[r] + bias0;
        out[obase + gc1] = acc1[r] + bias1;
    }
}

// ------------------------------------------------------------------
extern "C" void kernel_launch(void* const* d_in, const int* in_sizes, int n_in,
                              void* d_out, int out_size, void* d_ws, size_t ws_size,
                              hipStream_t stream) {
    const float* x        = (const float*)d_in[0];
    const float* H0       = (const float*)d_in[1];
    const float* W        = (const float*)d_in[2];
    const float* U        = (const float*)d_in[3];
    const float* hid_bias = (const float*)d_in[4];
    const float* V        = (const float*)d_in[5];
    const float* out_bias = (const float*)d_in[6];

    float* hidden = (float*)d_out;                       // (B,S,NH) fp32
    float* outp   = hidden + (size_t)B_ * S_ * NH_;      // (B,S,NO) fp32
    char*  ws     = (char*)d_ws;

    const size_t needXT   = kWS_XT + kXT_BYTES;
    const size_t needFull = kWS_HSEQ + kHSEQ_BYTES;
    const int mode = (ws_size >= needFull) ? 2 : (ws_size >= needXT) ? 1 : 0;

    __bf16* hseq  = (__bf16*)(ws + kWS_HSEQ);
    __bf16* h0dst = (mode == 2) ? hseq : (__bf16*)(ws + kWS_H0);

    const int prepTotal = 512 * KTOT_ + NO_ * NH_ + B_ * NH_ + 1;
    srn_prep<<<(prepTotal + 255) / 256, 256, 0, stream>>>(H0, W, U, V, ws, h0dst);

    if (mode >= 1) {
        const size_t totalX = (size_t)S_ * B_ * NI_;
        srn_xpose<<<(unsigned)((totalX + 255) / 256), 256, 0, stream>>>(x, ws);
    }

    const __bf16* xT = (const __bf16*)(ws + kWS_XT);
    switch (mode) {
    case 2:
        srn_recur<2><<<NWG_, 256, 64 * KTOT_ * 2, stream>>>(x, xT, hseq, hid_bias, ws, hidden);
        srn_out<true><<<dim3(S_, NO_ / 64), 256, 64 * NH_ * 2, stream>>>(hidden, out_bias, ws, outp);
        break;
    case 1:
        srn_recur<1><<<NWG_, 256, 64 * KTOT_ * 2, stream>>>(x, xT, nullptr, hid_bias, ws, hidden);
        srn_out<false><<<dim3((B_ * S_) / 64, NO_ / 64), 256, 64 * NH_ * 2, stream>>>(hidden, out_bias, ws, outp);
        break;
    default:
        srn_recur<0><<<NWG_, 256, 64 * KTOT_ * 2, stream>>>(x, nullptr, nullptr, hid_bias, ws, hidden);
        srn_out<false><<<dim3((B_ * S_) / 64, NO_ / 64), 256, 64 * NH_ * 2, stream>>>(hidden, out_bias, ws, outp);
        break;
    }
}
